// Graph2Seq_53326313947353
// MI455X (gfx1250) — matrix operand order repaired and checked by reading, not compiled, not run on hardware
//
#include <hip/hip_runtime.h>

// ---------------------------------------------------------------------------
// Graph2Seq (TSP pointer-net) for MI455X / gfx1250, wave32, WMMA bf16,
// TDM (tensor_load_to_lds) double-buffered GEMM + async-LDS attention staging.
// B=512, N=100, E=256, H=8, DH=32, HID=1024, L=6.
// ---------------------------------------------------------------------------

#define B_    512
#define N_    100
#define E_    256
#define H_    8
#define DH_   32
#define HID_  1024
#define L_    6
#define NPAD_ 128
#define NEG_  (-1e9f)

typedef __bf16 bf16_t;
typedef __attribute__((ext_vector_type(16))) __bf16    v16bf;
typedef __attribute__((ext_vector_type(8)))  __bf16    v8bf;
typedef __attribute__((ext_vector_type(8)))  float     v8f;
typedef __attribute__((ext_vector_type(4)))  int       v4i;
typedef __attribute__((ext_vector_type(4)))  unsigned  u32x4;
typedef __attribute__((ext_vector_type(8)))  int       i32x8;
typedef __attribute__((ext_vector_type(4)))  int       i32x4;

#if __has_builtin(__builtin_amdgcn_global_load_async_to_lds_b128)
#define HAS_ASYNC_LDS 1
#else
#define HAS_ASYNC_LDS 0
#endif
#if __has_builtin(__builtin_amdgcn_tensor_load_to_lds)
#define HAS_TDM 1
#else
#define HAS_TDM 0
#endif

// ---------------- async global->LDS helpers ---------------------------------

__device__ __forceinline__ void async_copy16(const bf16_t* g, bf16_t* l) {
#if HAS_ASYNC_LDS
  __builtin_amdgcn_global_load_async_to_lds_b128(
      (__attribute__((address_space(1))) v4i*)g,
      (__attribute__((address_space(3))) v4i*)l, 0, 0);
#else
  *(v8bf*)l = *(const v8bf*)g;
#endif
}

template <int CNT>
__device__ __forceinline__ void async_wait() {
#if HAS_ASYNC_LDS
#if __has_builtin(__builtin_amdgcn_s_wait_asynccnt)
  __builtin_amdgcn_s_wait_asynccnt((unsigned short)CNT);
#else
  asm volatile("s_wait_asynccnt %0" ::"i"(CNT) : "memory");
#endif
#endif
}

template <int CNT>
__device__ __forceinline__ void tensor_wait() {
#if HAS_TDM
#if __has_builtin(__builtin_amdgcn_s_wait_tensorcnt)
  __builtin_amdgcn_s_wait_tensorcnt((unsigned short)CNT);
#else
  asm volatile("s_wait_tensorcnt %0" ::"i"(CNT) : "memory");
#endif
#endif
}

// LDS byte offset of a __shared__ object (generic -> AS(3) -> int).
__device__ __forceinline__ unsigned lds_offset(const void* p) {
  __attribute__((address_space(3))) const void* lp =
      (__attribute__((address_space(3))) const void*)p;
  return (unsigned)(size_t)lp;
}

// ---------------- TDM: 2D bf16 tile load, global -> LDS ---------------------
// D# per CDNA5 ISA ch.8: group0 {count, lds_addr, global_addr, type=2},
// group1 {data_size=2B, pad ctrl, tensor dims, tile dims, dim0 stride}.
// pad_int_code: 2<<code dwords between pads; pad_amt_code: code+1 dwords pad.
__device__ __forceinline__ void tdm_load_2d(const void* gp, unsigned lds_off,
                                            int dim0, int dim1, int stride0,
                                            int tile0, int tile1,
                                            int pad_int_code, int pad_amt_code) {
#if HAS_TDM
  size_t ga = (size_t)gp;
  u32x4 g0;
  g0[0] = 1u;                                   // count=1, user descriptor
  g0[1] = lds_off;                              // LDS byte address
  g0[2] = (unsigned)(ga & 0xFFFFFFFFu);         // global_addr[31:0]
  g0[3] = (unsigned)((ga >> 32) & 0x1FFFFFFu)   // global_addr[56:32]
          | (2u << 30);                         // type=2 ("image")
  i32x8 g1;
  g1[0] = (int)((1u << 16)                      // data_size = 2 bytes
                | (1u << 20)                    // pad_enable
                | ((unsigned)pad_int_code << 22)
                | ((unsigned)pad_amt_code << 25));
  g1[1] = (int)(((unsigned)dim0 & 0xFFFFu) << 16);              // tensor_dim0 lo
  g1[2] = (int)((((unsigned)dim0 >> 16) & 0xFFFFu)
                | (((unsigned)dim1 & 0xFFFFu) << 16));          // dim0 hi | dim1 lo
  g1[3] = (int)((((unsigned)dim1 >> 16) & 0xFFFFu)
                | ((unsigned)tile0 << 16));                     // dim1 hi | tile0
  g1[4] = (int)((unsigned)tile1 & 0xFFFFu);                     // tile1 | tile2=0
  g1[5] = stride0;                                              // dim0 stride lo
  g1[6] = 0;                                                    // stride hi | dim1 stride
  g1[7] = 0;
  i32x4 z4 = {0, 0, 0, 0};
#if __clang_major__ >= 23
  i32x8 z8 = {0, 0, 0, 0, 0, 0, 0, 0};
  __builtin_amdgcn_tensor_load_to_lds(g0, g1, z4, z4, z8, 0);
#else
  __builtin_amdgcn_tensor_load_to_lds(g0, g1, z4, z4, 0);
#endif
#else
  (void)gp; (void)lds_off; (void)dim0; (void)dim1; (void)stride0;
  (void)tile0; (void)tile1; (void)pad_int_code; (void)pad_amt_code;
#endif
}

// ---------------- WMMA fragment helpers (CDNA5 ISA 7.12.2 layouts) ----------

__device__ __forceinline__ v16bf frag_cat(v8bf lo, v8bf hi) {
  v16bf f;
#pragma unroll
  for (int i = 0; i < 8; ++i) { f[i] = lo[i]; f[i + 8] = hi[i]; }
  return f;
}

// A matrix 16x32 (MxK): lane -> mrow=lane&15, khalf=lane>>4.
__device__ __forceinline__ v16bf load_fragA(const bf16_t* base, int ld,
                                            int mrow, int khalf) {
  const bf16_t* r = base + (size_t)mrow * ld + khalf * 8;
  v8bf lo = *(const v8bf*)(r);
  v8bf hi = *(const v8bf*)(r + 16);
  return frag_cat(lo, hi);
}

// B matrix 32x16 (KxN): lane holds column n0+(kr&15), K rows 8*(kr>>4)+s and 16+8*(kr>>4)+s.
__device__ __forceinline__ v16bf load_fragB(const bf16_t* base, int ld,
                                            int kr, int n0) {
  const bf16_t* r = base + (size_t)((kr >> 4) * 8) * ld + n0 + (kr & 15);
  v16bf f;
#pragma unroll
  for (int i = 0; i < 8; ++i) {
    f[i] = r[(size_t)i * ld];
    f[i + 8] = r[(size_t)(16 + i) * ld];
  }
  return f;
}

#define WMMA_BF16(a, b, c) \
  __builtin_amdgcn_wmma_f32_16x16x32_bf16(false, (a), false, (b), (short)0, (c), false, false)

__device__ __forceinline__ float wave_sum(float v) {
#pragma unroll
  for (int o = 16; o > 0; o >>= 1) v += __shfl_xor(v, o, 32);
  return v;
}
__device__ __forceinline__ float wave_max(float v) {
#pragma unroll
  for (int o = 16; o > 0; o >>= 1) v = fmaxf(v, __shfl_xor(v, o, 32));
  return v;
}

// ---------------- fp32 -> bf16 conversion -----------------------------------

__global__ __launch_bounds__(256) void f2bf_kernel(const float* __restrict__ src,
                                                   bf16_t* __restrict__ dst, int n) {
  int i = blockIdx.x * 256 + threadIdx.x;
  if (i < n) dst[i] = (bf16_t)src[i];
}

// ---------------- input embedding: h = x @ Win + bin ------------------------

__global__ __launch_bounds__(256) void embed_kernel(const float* __restrict__ x,
                                                    const float* __restrict__ Win,
                                                    const float* __restrict__ bin,
                                                    float* __restrict__ hF,
                                                    bf16_t* __restrict__ hB) {
  const int row = blockIdx.x;          // 0..B*N-1
  const int e = threadIdx.x;           // 0..E-1
  float x0 = x[(size_t)row * 2 + 0];
  float x1 = x[(size_t)row * 2 + 1];
  float v = x0 * Win[e] + x1 * Win[E_ + e] + bin[e];
  size_t off = (size_t)row * E_ + e;
  hF[off] = v;
  hB[off] = (bf16_t)v;
}

// ---------------- generic tiled bf16 WMMA GEMM ------------------------------
// C[M,N] = A[M,K] @ W[K,N] (+bias) (+relu). M%128==0, N%128==0, K%32==0.
// Block: 256 threads = 8 waves (4 along M x 2 along N), 32x64 f32 acc / wave.
// Double-buffered LDS; tiles fetched by the Tensor Data Mover (wave 0 issues
// two 2D descriptors per K-step; TENSORcnt pipeline), HW pad matches LDS pad.

__global__ __launch_bounds__(256) void gemm_bf16_wmma(
    const bf16_t* __restrict__ A, const bf16_t* __restrict__ W,
    const float* __restrict__ bias, float* __restrict__ Cf,
    bf16_t* __restrict__ Cb, int M, int N, int K, int relu) {
  __shared__ __align__(16) bf16_t As[2][128][40];   // 32 halves + 8 pad / row
  __shared__ __align__(16) bf16_t Ws[2][32][136];   // 128 halves + 8 pad / row
  const int tid = threadIdx.x;
  const int wave = tid >> 5, lane = tid & 31;
  const int wm = wave & 3, wn = wave >> 2;          // 4 x 2 wave grid
  const int mrow = lane & 15, khalf = lane >> 4;
  const int kr = mrow + 16 * khalf;
  const int m0 = blockIdx.y * 128, n0 = blockIdx.x * 128;
  const int nk = K >> 5;
  v8f acc[2][4] = {};

  auto issue = [&](int k0, int bufi) {
#if HAS_TDM
    if (wave == 0) {
      // A tile: 128 rows x 32 halves, row pitch K; pad 16dw rows by 4dw.
      tdm_load_2d(A + (size_t)m0 * K + k0, lds_offset(&As[bufi][0][0]),
                  K - k0, M - m0, K, 32, 128, /*pad 2<<3=16dw*/3, /*4dw*/3);
      // W tile: 32 rows x 128 halves, row pitch N; pad 64dw rows by 4dw.
      tdm_load_2d(W + (size_t)k0 * N + n0, lds_offset(&Ws[bufi][0][0]),
                  N - n0, K - k0, N, 128, 32, /*pad 2<<5=64dw*/5, /*4dw*/3);
    }
#else
    const int ar = tid >> 1, ac = (tid & 1) * 16;
    const int wr = tid >> 3, wc = (tid & 7) * 16;
    const bf16_t* ga = A + (size_t)(m0 + ar) * K + k0 + ac;
    const bf16_t* gw = W + (size_t)(k0 + wr) * N + n0 + wc;
    async_copy16(ga,     &As[bufi][ar][ac]);
    async_copy16(ga + 8, &As[bufi][ar][ac + 8]);
    async_copy16(gw,     &Ws[bufi][wr][wc]);
    async_copy16(gw + 8, &Ws[bufi][wr][wc + 8]);
#endif
  };

  issue(0, 0);
  for (int t = 0; t < nk; ++t) {
    const int cur = t & 1;
    if (t + 1 < nk) {
      issue((t + 1) << 5, cur ^ 1);
#if HAS_TDM
      tensor_wait<2>();   // tile t's two TDM descriptors have completed
#else
      async_wait<4>();
#endif
    } else {
#if HAS_TDM
      tensor_wait<0>();
#else
      async_wait<0>();
#endif
    }
    __syncthreads();      // all waves see tile t in LDS
    const bf16_t (*Ab)[40]  = As[cur];
    const bf16_t (*Wb)[136] = Ws[cur];
    v16bf a0 = load_fragA(&Ab[wm * 32][0], 40, mrow, khalf);
    v16bf a1 = load_fragA(&Ab[wm * 32 + 16][0], 40, mrow, khalf);
#pragma unroll
    for (int j = 0; j < 4; ++j) {
      v16bf bj = load_fragB(&Wb[0][0], 136, kr, wn * 64 + j * 16);
      acc[0][j] = WMMA_BF16(a0, bj, acc[0][j]);
      acc[1][j] = WMMA_BF16(a1, bj, acc[1][j]);
    }
    __syncthreads();      // done reading buf[cur]; next issue may overwrite
  }

#pragma unroll
  for (int i = 0; i < 2; ++i)
#pragma unroll
    for (int j = 0; j < 4; ++j) {
      int col = n0 + wn * 64 + j * 16 + mrow;
      float bv = bias ? bias[col] : 0.f;
#pragma unroll
      for (int v = 0; v < 8; ++v) {
        int row = m0 + wm * 32 + i * 16 + v + 8 * khalf;
        float val = acc[i][j][v] + bv;
        if (relu) val = fmaxf(val, 0.f);
        size_t off = (size_t)row * N + col;
        if (Cf) Cf[off] = val;
        if (Cb) Cb[off] = (bf16_t)val;
      }
    }
}

// ---------------- fused encoder self-attention (one block per (b,h)) --------
// S = (Q K^T)/sqrt(DH) via WMMA; fp32 softmax; O = P V via WMMA.
// LDS < 64KB via two 64-row halves; bf16 P aliases the fp32 score buffer.

__global__ __launch_bounds__(256) void enc_attn_kernel(
    const bf16_t* __restrict__ qB, const bf16_t* __restrict__ kB,
    const bf16_t* __restrict__ vB, bf16_t* __restrict__ aoB) {
  __shared__ __align__(16) bf16_t Qs[NPAD_][40];
  __shared__ __align__(16) bf16_t Kts[DH_][NPAD_ + 8];
  __shared__ __align__(16) bf16_t Vs[NPAD_][40];
  __shared__ __align__(16) float Ss[64][NPAD_ + 8];  // fp32 scores (one half)
  const int b = blockIdx.x / H_, h = blockIdx.x % H_;
  const int tid = threadIdx.x, wave = tid >> 5, lane = tid & 31;
  const int mrow = lane & 15, khalf = lane >> 4;
  const int kr = mrow + 16 * khalf;
  const size_t base = (size_t)b * N_ * E_ + h * DH_;
  const float scale = 0.17677669529663687f;  // 1/sqrt(32)
  bf16_t* Ps = (bf16_t*)&Ss[0][0];
  const int ldp = (NPAD_ + 8) * 2;  // 272 halves per P row

  {  // stage Q, V row-major (async copies); K transposed [d][n] (sync)
    const int row = tid >> 1, ch = (tid & 1) * 16;
    if (row < N_) {
      const bf16_t* gq = qB + base + (size_t)row * E_ + ch;
      const bf16_t* gv = vB + base + (size_t)row * E_ + ch;
      async_copy16(gq,     &Qs[row][ch]);
      async_copy16(gq + 8, &Qs[row][ch + 8]);
      async_copy16(gv,     &Vs[row][ch]);
      async_copy16(gv + 8, &Vs[row][ch + 8]);
      const bf16_t* gk = kB + base + (size_t)row * E_ + ch;
      v8bf k0 = *(const v8bf*)(gk);
      v8bf k1 = *(const v8bf*)(gk + 8);
#pragma unroll
      for (int j = 0; j < 8; ++j) {
        Kts[ch + j][row] = k0[j];
        Kts[ch + 8 + j][row] = k1[j];
      }
    } else {
      v8bf z = {};
      *(v8bf*)&Qs[row][ch] = z;
      *(v8bf*)&Qs[row][ch + 8] = z;
      *(v8bf*)&Vs[row][ch] = z;
      *(v8bf*)&Vs[row][ch + 8] = z;
#pragma unroll
      for (int j = 0; j < 16; ++j) Kts[ch + j][row] = (bf16_t)0.f;
    }
    async_wait<0>();
  }

  for (int half = 0; half < 2; ++half) {
    const int mbase = half * 64;
    __syncthreads();  // staging visibility + Ss reuse guard
    // S tiles: 4 (M) x 8 (N) = 32 tiles, one WMMA each (K = DH = 32)
    for (int t = wave; t < 32; t += 8) {
      int ti = t >> 3, tj = t & 7;
      v16bf a = load_fragA(&Qs[mbase + ti * 16][0], 40, mrow, khalf);
      v16bf bb = load_fragB(&Kts[0][0], NPAD_ + 8, kr, tj * 16);
      v8f c = {};
      c = WMMA_BF16(a, bb, c);
#pragma unroll
      for (int v = 0; v < 8; ++v)
        Ss[ti * 16 + v + 8 * khalf][tj * 16 + mrow] = c[v] * scale;
    }
    __syncthreads();
    // fp32 softmax per row -> bf16 P (in place over Ss)
    for (int r = wave; r < 64; r += 8) {
      int gr = mbase + r;
      if (gr >= N_) continue;
      float sv[4];
#pragma unroll
      for (int i = 0; i < 4; ++i) {
        int c = lane + 32 * i;
        sv[i] = (c < N_) ? Ss[r][c] : -3.0e38f;
      }
      float mx = wave_max(fmaxf(fmaxf(sv[0], sv[1]), fmaxf(sv[2], sv[3])));
      float ev[4], s = 0.f;
#pragma unroll
      for (int i = 0; i < 4; ++i) {
        int c = lane + 32 * i;
        ev[i] = (c < N_) ? __expf(sv[i] - mx) : 0.f;
        s += ev[i];
      }
      s = wave_sum(s);
      float inv = 1.f / s;
#pragma unroll
      for (int i = 0; i < 4; ++i) {  // all reads done (regs); safe to alias
        int c = lane + 32 * i;
        Ps[(size_t)r * ldp + c] = (bf16_t)(ev[i] * inv);
      }
    }
    __syncthreads();
    // O = P V : 4 (M) x 2 (DH) tiles, k-loop over 4 WMMA steps
    {
      int ti = wave >> 1, dj = (wave & 1) * 16;
      v8f c = {};
#pragma unroll
      for (int kk = 0; kk < 4; ++kk) {
        v16bf a = load_fragA(Ps + (size_t)(ti * 16) * ldp + kk * 32, ldp, mrow, khalf);
        v16bf bb = load_fragB(&Vs[kk * 32][0], 40, kr, dj);
        c = WMMA_BF16(a, bb, c);
      }
#pragma unroll
      for (int v = 0; v < 8; ++v) {
        int gr = mbase + ti * 16 + v + 8 * khalf;
        int col = dj + mrow;
        if (gr < N_) aoB[base + (size_t)gr * E_ + col] = (bf16_t)c[v];
      }
    }
  }
}

// ---------------- residual add + LayerNorm (fp32 master) --------------------

__global__ __launch_bounds__(256) void add_ln_kernel(
    const float* __restrict__ hin, const float* __restrict__ delta,
    const float* __restrict__ lns, const float* __restrict__ lnb,
    float* __restrict__ hF, bf16_t* __restrict__ hB) {
  const int row = blockIdx.x, tid = threadIdx.x;
  size_t off = (size_t)row * E_ + tid;
  float x = hin[off] + delta[off];
  float s1 = wave_sum(x), s2 = wave_sum(x * x);
  __shared__ float r1[8], r2[8], mrs[2];
  if ((tid & 31) == 0) { r1[tid >> 5] = s1; r2[tid >> 5] = s2; }
  __syncthreads();
  if (tid == 0) {
    float a = 0.f, bb = 0.f;
#pragma unroll
    for (int i = 0; i < 8; ++i) { a += r1[i]; bb += r2[i]; }
    float mean = a * (1.f / E_);
    float var = bb * (1.f / E_) - mean * mean;
    mrs[0] = mean; mrs[1] = rsqrtf(var + 1e-5f);
  }
  __syncthreads();
  float y = (x - mrs[0]) * mrs[1] * lns[tid] + lnb[tid];
  hF[off] = y;
  hB[off] = (bf16_t)y;
}

// ---------------- graph mean + initial decoder context ----------------------

__global__ __launch_bounds__(256) void graph_ctx_kernel(
    const float* __restrict__ hF, const float* __restrict__ tok1,
    const float* __restrict__ tokf, float* __restrict__ graphF,
    float* __restrict__ ctxF, bf16_t* __restrict__ ctxB) {
  const int b = blockIdx.x, e = threadIdx.x;
  float s = 0.f;
  for (int n = 0; n < N_; ++n) s += hF[((size_t)b * N_ + n) * E_ + e];
  s *= (1.f / N_);
  graphF[(size_t)b * E_ + e] = s;
  size_t cb = (size_t)b * 3 * E_;
  float t1 = tok1[e], tf = tokf[e];
  ctxF[cb + e] = s;           ctxB[cb + e] = (bf16_t)s;
  ctxF[cb + E_ + e] = t1;     ctxB[cb + E_ + e] = (bf16_t)t1;
  ctxF[cb + 2 * E_ + e] = tf; ctxB[cb + 2 * E_ + e] = (bf16_t)tf;
}

// ---------------- decoder: masked MHA over nodes (one block per b) ----------

__global__ __launch_bounds__(256) void dec_attn_kernel(
    const float* __restrict__ qhF, const bf16_t* __restrict__ kD,
    const bf16_t* __restrict__ vD, const float* __restrict__ maskF,
    bf16_t* __restrict__ gB) {
  const int b = blockIdx.x, tid = threadIdx.x, h = tid >> 5, lane = tid & 31;
  __shared__ float qsh[E_];
  __shared__ float ps[H_][NPAD_];
  qsh[tid] = qhF[(size_t)b * E_ + tid];
  __syncthreads();
  const float scale = 0.17677669529663687f;
  float sc[4];
#pragma unroll
  for (int i = 0; i < 4; ++i) {
    int n = lane + 32 * i;
    float s = -3.0e38f;
    if (n < N_) {
      const bf16_t* kp = kD + ((size_t)b * N_ + n) * E_ + h * DH_;
      float acc = 0.f;
#pragma unroll
      for (int d = 0; d < DH_; ++d) acc += qsh[h * DH_ + d] * (float)kp[d];
      s = acc * scale;
      if (maskF[(size_t)b * N_ + n] > 0.f) s = NEG_;
    }
    sc[i] = s;
  }
  float mx = wave_max(fmaxf(fmaxf(sc[0], sc[1]), fmaxf(sc[2], sc[3])));
  float ev[4], s = 0.f;
#pragma unroll
  for (int i = 0; i < 4; ++i) {
    int n = lane + 32 * i;
    ev[i] = (n < N_) ? __expf(sc[i] - mx) : 0.f;
    s += ev[i];
  }
  s = wave_sum(s);
  float inv = 1.f / s;
#pragma unroll
  for (int i = 0; i < 4; ++i) ps[h][lane + 32 * i] = ev[i] * inv;
  __syncthreads();
  {
    int d = lane;
    const bf16_t* vp = vD + (size_t)b * N_ * E_ + h * DH_ + d;
    float acc = 0.f;
    for (int n = 0; n < N_; ++n) acc += ps[h][n] * (float)vp[(size_t)n * E_];
    gB[(size_t)b * E_ + h * DH_ + d] = (bf16_t)acc;
  }
}

// ---------------- decoder: pointer logits/softmax/argmax/ctx update ---------

__global__ __launch_bounds__(256) void dec_point_kernel(
    const float* __restrict__ goutF, const float* __restrict__ kpF,
    const float* __restrict__ hF, float* __restrict__ maskF,
    float* __restrict__ startF, float* __restrict__ ctxF,
    bf16_t* __restrict__ ctxB, float* __restrict__ probsOut,
    float* __restrict__ toursOut, int step) {
  const int b = blockIdx.x, tid = threadIdx.x, wave = tid >> 5, lane = tid & 31;
  __shared__ float gsh[E_];
  __shared__ float lg[NPAD_];
  __shared__ int sidx;
  gsh[tid] = goutF[(size_t)b * E_ + tid];
  __syncthreads();
  for (int n = wave; n < NPAD_; n += 8) {
    float l = -3.0e38f;
    if (n < N_) {
      const float* kp = kpF + ((size_t)b * N_ + n) * E_;
      float acc = 0.f;
      for (int d = lane; d < E_; d += 32) acc += gsh[d] * kp[d];
      acc = wave_sum(acc);
      l = 10.f * tanhf(acc * 0.0625f);  // /sqrt(E)=1/16
      if (maskF[(size_t)b * N_ + n] > 0.f) l = NEG_;
    }
    if (lane == 0) lg[n] = l;
  }
  __syncthreads();
  if (wave == 0) {
    float mv[4];
    float mx = -3.0e38f;
#pragma unroll
    for (int i = 0; i < 4; ++i) {
      int n = lane + 32 * i;
      mv[i] = lg[n];
      if (n < N_) mx = fmaxf(mx, mv[i]);
    }
    mx = wave_max(mx);
    float ev[4], s = 0.f;
#pragma unroll
    for (int i = 0; i < 4; ++i) {
      int n = lane + 32 * i;
      ev[i] = (n < N_) ? __expf(mv[i] - mx) : 0.f;
      s += ev[i];
    }
    s = wave_sum(s);
    float inv = 1.f / s;
#pragma unroll
    for (int i = 0; i < 4; ++i) {
      int n = lane + 32 * i;
      if (n < N_)
        probsOut[((size_t)b * N_ + step) * N_ + n] = ev[i] * inv;
    }
    // argmax (first occurrence of max)
    float bm = -3.0e38f; int bi = 0x7fffffff;
#pragma unroll
    for (int i = 0; i < 4; ++i) {
      int n = lane + 32 * i;
      if (n < N_ && mv[i] > bm) { bm = mv[i]; bi = n; }
    }
#pragma unroll
    for (int o = 16; o > 0; o >>= 1) {
      float obm = __shfl_xor(bm, o, 32);
      int obi = __shfl_xor(bi, o, 32);
      if (obm > bm || (obm == bm && obi < bi)) { bm = obm; bi = obi; }
    }
    if (lane == 0) {
      sidx = bi;
      toursOut[(size_t)b * N_ + step] = (float)bi;
      maskF[(size_t)b * N_ + bi] = 1.f;
    }
  }
  __syncthreads();
  const int idx = sidx;
  float sel = hF[((size_t)b * N_ + idx) * E_ + tid];
  float st;
  if (step == 0) { startF[(size_t)b * E_ + tid] = sel; st = sel; }
  else           { st = startF[(size_t)b * E_ + tid]; }
  size_t cb = (size_t)b * 3 * E_;
  ctxF[cb + E_ + tid] = st;       ctxB[cb + E_ + tid] = (bf16_t)st;
  ctxF[cb + 2 * E_ + tid] = sel;  ctxB[cb + 2 * E_ + tid] = (bf16_t)sel;
}

// ---------------- NLL (deterministic 2-stage reduction) ---------------------

__global__ __launch_bounds__(128) void nll_part_kernel(
    const float* __restrict__ probs, const int* __restrict__ target,
    float* __restrict__ part) {
  const int b = blockIdx.x, tid = threadIdx.x;
  float s = 0.f;
  for (int n = tid; n < N_; n += 128) {
    int tg = target[(size_t)b * N_ + n];
    float p = probs[((size_t)b * N_ + n) * N_ + tg];
    s += __logf(p + 1e-9f);
  }
  s = wave_sum(s);
  __shared__ float r[4];
  if ((tid & 31) == 0) r[tid >> 5] = s;
  __syncthreads();
  if (tid == 0) part[b] = r[0] + r[1] + r[2] + r[3];
}

__global__ __launch_bounds__(256) void nll_final_kernel(
    const float* __restrict__ part, float* __restrict__ out) {
  const int tid = threadIdx.x;
  float s = 0.f;
  for (int b = tid; b < B_; b += 256) s += part[b];
  s = wave_sum(s);
  __shared__ float r[8];
  if ((tid & 31) == 0) r[tid >> 5] = s;
  __syncthreads();
  if (tid == 0) {
    float t = 0.f;
#pragma unroll
    for (int i = 0; i < 8; ++i) t += r[i];
    out[0] = -t / (float)(B_ * N_);
  }
}

// ---------------------------------------------------------------------------
// Host orchestration
// ---------------------------------------------------------------------------

extern "C" void kernel_launch(void* const* d_in, const int* in_sizes, int n_in,
                              void* d_out, int out_size, void* d_ws, size_t ws_size,
                              hipStream_t stream) {
  (void)in_sizes; (void)n_in; (void)out_size; (void)ws_size;
  const float* x       = (const float*)d_in[0];
  const int*   target  = (const int*)  d_in[1];
  const float* enc_Win = (const float*)d_in[2];
  const float* enc_bin = (const float*)d_in[3];
  const float* enc_Wq  = (const float*)d_in[4];
  const float* enc_Wk  = (const float*)d_in[5];
  const float* enc_Wv  = (const float*)d_in[6];
  const float* enc_Wo  = (const float*)d_in[7];
  const float* ln1s    = (const float*)d_in[8];
  const float* ln1b    = (const float*)d_in[9];
  const float* enc_W1  = (const float*)d_in[10];
  const float* enc_b1  = (const float*)d_in[11];
  const float* enc_W2  = (const float*)d_in[12];
  const float* enc_b2  = (const float*)d_in[13];
  const float* ln2s    = (const float*)d_in[14];
  const float* ln2b    = (const float*)d_in[15];
  const float* dWq     = (const float*)d_in[16];
  const float* dWk     = (const float*)d_in[17];
  const float* dWv     = (const float*)d_in[18];
  const float* dWo     = (const float*)d_in[19];
  const float* dWp     = (const float*)d_in[20];
  const float* tok1    = (const float*)d_in[21];
  const float* tokf    = (const float*)d_in[22];

  const size_t BN = (size_t)B_ * N_;

  // ---- workspace bump allocator ----
  char* wp = (char*)d_ws;
  auto alloc = [&](size_t bytes) -> void* {
    void* r = (void*)wp;
    wp += (bytes + 255) & ~(size_t)255;
    return r;
  };
  bf16_t* WqB  = (bf16_t*)alloc(sizeof(bf16_t) * L_ * E_ * E_);
  bf16_t* WkB  = (bf16_t*)alloc(sizeof(bf16_t) * L_ * E_ * E_);
  bf16_t* WvB  = (bf16_t*)alloc(sizeof(bf16_t) * L_ * E_ * E_);
  bf16_t* WoB  = (bf16_t*)alloc(sizeof(bf16_t) * L_ * E_ * E_);
  bf16_t* W1B  = (bf16_t*)alloc(sizeof(bf16_t) * L_ * E_ * HID_);
  bf16_t* W2B  = (bf16_t*)alloc(sizeof(bf16_t) * L_ * HID_ * E_);
  bf16_t* dWqB = (bf16_t*)alloc(sizeof(bf16_t) * 3 * E_ * E_);
  bf16_t* dWkB = (bf16_t*)alloc(sizeof(bf16_t) * E_ * E_);
  bf16_t* dWvB = (bf16_t*)alloc(sizeof(bf16_t) * E_ * E_);
  bf16_t* dWoB = (bf16_t*)alloc(sizeof(bf16_t) * E_ * E_);
  bf16_t* dWpB = (bf16_t*)alloc(sizeof(bf16_t) * E_ * E_);
  float*  hF     = (float*) alloc(sizeof(float)  * BN * E_);
  bf16_t* hB     = (bf16_t*)alloc(sizeof(bf16_t) * BN * E_);
  bf16_t* qBbuf  = (bf16_t*)alloc(sizeof(bf16_t) * BN * E_);
  bf16_t* kBbuf  = (bf16_t*)alloc(sizeof(bf16_t) * BN * E_);
  bf16_t* vBbuf  = (bf16_t*)alloc(sizeof(bf16_t) * BN * E_);
  bf16_t* aoB    = (bf16_t*)alloc(sizeof(bf16_t) * BN * E_);
  float*  deltaF = (float*) alloc(sizeof(float)  * BN * E_);
  bf16_t* ffhB   = (bf16_t*)alloc(sizeof(bf16_t) * BN * HID_);
  float*  kpF    = (float*) alloc(sizeof(float)  * BN * E_);
  float*  graphF = (float*) alloc(sizeof(float)  * B_ * E_);
  float*  ctxF   = (float*) alloc(sizeof(float)  * B_ * 3 * E_);
  bf16_t* ctxB   = (bf16_t*)alloc(sizeof(bf16_t) * B_ * 3 * E_);
  float*  qhF    = (float*) alloc(sizeof(float)  * B_ * E_);
  bf16_t* gB     = (bf16_t*)alloc(sizeof(bf16_t) * B_ * E_);
  float*  goutF  = (float*) alloc(sizeof(float)  * B_ * E_);
  float*  maskF  = (float*) alloc(sizeof(float)  * B_ * N_);
  float*  startF = (float*) alloc(sizeof(float)  * B_ * E_);
  float*  partF  = (float*) alloc(sizeof(float)  * B_);

  float* probsOut = (float*)d_out;                    // [B,N,N]
  float* toursOut = probsOut + (size_t)B_ * N_ * N_;  // [B,N]
  float* nllOut   = toursOut + (size_t)B_ * N_;       // [1]

  auto cvt = [&](const float* src, bf16_t* dst, size_t n) {
    f2bf_kernel<<<dim3((unsigned)((n + 255) / 256)), dim3(256), 0, stream>>>(src, dst, (int)n);
  };
  auto gemm = [&](const bf16_t* A, const bf16_t* W, const float* bias,
                  float* Cf, bf16_t* Cb, int M, int Nn, int K, int relu) {
    gemm_bf16_wmma<<<dim3(Nn / 128, M / 128), dim3(256), 0, stream>>>(
        A, W, bias, Cf, Cb, M, Nn, K, relu);
  };

  // ---- weight conversion to bf16 ----
  cvt(enc_Wq, WqB, (size_t)L_ * E_ * E_);
  cvt(enc_Wk, WkB, (size_t)L_ * E_ * E_);
  cvt(enc_Wv, WvB, (size_t)L_ * E_ * E_);
  cvt(enc_Wo, WoB, (size_t)L_ * E_ * E_);
  cvt(enc_W1, W1B, (size_t)L_ * E_ * HID_);
  cvt(enc_W2, W2B, (size_t)L_ * HID_ * E_);
  cvt(dWq, dWqB, (size_t)3 * E_ * E_);
  cvt(dWk, dWkB, (size_t)E_ * E_);
  cvt(dWv, dWvB, (size_t)E_ * E_);
  cvt(dWo, dWoB, (size_t)E_ * E_);
  cvt(dWp, dWpB, (size_t)E_ * E_);

  // ---- encoder ----
  embed_kernel<<<dim3((unsigned)BN), dim3(E_), 0, stream>>>(x, enc_Win, enc_bin, hF, hB);
  for (int l = 0; l < L_; ++l) {
    const size_t wo = (size_t)l * E_ * E_;
    gemm(hB, WqB + wo, nullptr, nullptr, qBbuf, (int)BN, E_, E_, 0);
    gemm(hB, WkB + wo, nullptr, nullptr, kBbuf, (int)BN, E_, E_, 0);
    gemm(hB, WvB + wo, nullptr, nullptr, vBbuf, (int)BN, E_, E_, 0);
    enc_attn_kernel<<<dim3(B_ * H_), dim3(256), 0, stream>>>(qBbuf, kBbuf, vBbuf, aoB);
    gemm(aoB, WoB + wo, nullptr, deltaF, nullptr, (int)BN, E_, E_, 0);
    add_ln_kernel<<<dim3((unsigned)BN), dim3(E_), 0, stream>>>(
        hF, deltaF, ln1s + (size_t)l * E_, ln1b + (size_t)l * E_, hF, hB);
    gemm(hB, W1B + (size_t)l * E_ * HID_, enc_b1 + (size_t)l * HID_,
         nullptr, ffhB, (int)BN, HID_, E_, 1);
    gemm(ffhB, W2B + (size_t)l * HID_ * E_, enc_b2 + (size_t)l * E_,
         deltaF, nullptr, (int)BN, E_, HID_, 0);
    add_ln_kernel<<<dim3((unsigned)BN), dim3(E_), 0, stream>>>(
        hF, deltaF, ln2s + (size_t)l * E_, ln2b + (size_t)l * E_, hF, hB);
  }

  // ---- decoder precompute ----
  graph_ctx_kernel<<<dim3(B_), dim3(E_), 0, stream>>>(hF, tok1, tokf, graphF, ctxF, ctxB);
  gemm(hB, dWkB, nullptr, nullptr, kBbuf, (int)BN, E_, E_, 0);  // decoder K (bf16)
  gemm(hB, dWvB, nullptr, nullptr, vBbuf, (int)BN, E_, E_, 0);  // decoder V (bf16)
  gemm(hB, dWpB, nullptr, kpF, nullptr, (int)BN, E_, E_, 0);    // pointer keys (f32)
  (void)hipMemsetAsync(maskF, 0, sizeof(float) * B_ * N_, stream);

  // ---- sequential greedy decode ----
  for (int s = 0; s < N_; ++s) {
    gemm(ctxB, dWqB, nullptr, qhF, nullptr, B_, E_, 3 * E_, 0);
    dec_attn_kernel<<<dim3(B_), dim3(256), 0, stream>>>(qhF, kBbuf, vBbuf, maskF, gB);
    gemm(gB, dWoB, nullptr, goutF, nullptr, B_, E_, E_, 0);
    dec_point_kernel<<<dim3(B_), dim3(256), 0, stream>>>(
        goutF, kpF, hF, maskF, startF, ctxF, ctxB, probsOut, toursOut, s);
  }

  // ---- NLL ----
  nll_part_kernel<<<dim3(B_), dim3(128), 0, stream>>>(probsOut, target, partF);
  nll_final_kernel<<<dim3(1), dim3(256), 0, stream>>>(partF, nllOut);
}
